// HSTU_BSA_Triton_30288109372030
// MI455X (gfx1250) — matrix operand order repaired and checked by reading, not compiled, not run on hardware
//
#include <hip/hip_runtime.h>
#include <hip/hip_bf16.h>

typedef __attribute__((ext_vector_type(16))) __bf16 v16bf;
typedef __attribute__((ext_vector_type(8)))  __bf16 v8bf;
typedef __attribute__((ext_vector_type(8)))  float  v8f;

#define T_TOK 3200
#define H     8
#define D     128
#define BATCH 4
#define NPAD  1024
#define BS    32
#define NB    32
#define STOPK 8
#define MT    64          /* NPAD/16 query tiles */
#define HD    1024        /* H*D */
#define SCALE 0.08838834764831845f   /* D^-0.5 */
#define NEGINF (-__builtin_inff())

__device__ __forceinline__ float siluf(float x) { return x / (1.f + __expf(-x)); }

// Build a 16-element bf16 A/B fragment from two 8-element chunks (contiguous k order:
// lanes 0-15 hold K{0..15}, lanes 16-31 hold K{16..31}).
__device__ __forceinline__ v16bf ld16(const __bf16* lo, const __bf16* hi) {
  v8bf a = *(const v8bf*)lo;
  v8bf b = *(const v8bf*)hi;
  v16bf r;
#pragma unroll
  for (int i = 0; i < 8; ++i) { r[i] = a[i]; r[i + 8] = b[i]; }
  return r;
}

__device__ __forceinline__ v8f wmma_bf16(v16bf a, v16bf b, v8f c) {
  return __builtin_amdgcn_wmma_f32_16x16x32_bf16(false, a, false, b, (short)0, c,
                                                 false, false);
}

// ---------------- K1: pad + fp32->bf16 (q,k) + gate MLP ----------------
__global__ void k_prep(const float* __restrict__ q, const float* __restrict__ k,
                       const float* __restrict__ Wg, const float* __restrict__ bg,
                       const int* __restrict__ off, __bf16* __restrict__ qbf,
                       __bf16* __restrict__ kbf, float* __restrict__ gcmp,
                       float* __restrict__ gslc) {
  int wid  = (blockIdx.x * blockDim.x + threadIdx.x) >> 5;  // one wave per (b,m,h)
  int lane = threadIdx.x & 31;
  int b = wid / (NPAD * H);
  int rem = wid % (NPAD * H);
  int m = rem / H;
  int h = rem % H;
  int o0 = off[b];
  int len = off[b + 1] - o0;
  bool valid = m < len;
  long gsrc = (long)(o0 + m) * HD + h * D;
  long gdst = (((long)b * H + h) * NPAD + m) * D;
  float s0 = 0.f, s1 = 0.f;
#pragma unroll
  for (int c = 0; c < 4; ++c) {
    int d = lane + 32 * c;
    float qq = valid ? q[gsrc + d] : 0.f;
    float kk = valid ? k[gsrc + d] : 0.f;
    qbf[gdst + d] = (__bf16)qq;
    kbf[gdst + d] = (__bf16)kk;
    const float* wg = Wg + (h * D + d) * 3;
    s0 += qq * wg[0];
    s1 += qq * wg[1];
  }
#pragma unroll
  for (int s = 16; s >= 1; s >>= 1) {
    s0 += __shfl_xor(s0, s, 32);
    s1 += __shfl_xor(s1, s, 32);
  }
  if (lane == 0) {
    long gi = ((long)b * H + h) * NPAD + m;
    gcmp[gi] = 1.f / (1.f + __expf(-(s0 + bg[h * 3 + 0])));
    gslc[gi] = 1.f / (1.f + __expf(-(s1 + bg[h * 3 + 1])));
  }
}

// ---------------- K1b: V transpose via LDS tile (coalesced both sides) --------
// vT[b,h,d,m] <- v[tok,h,d]; 32 m-rows x 128 d-cols per block.
__global__ void k_vtrans(const float* __restrict__ v, const int* __restrict__ off,
                         __bf16* __restrict__ vT) {
  __shared__ alignas(32) __bf16 tile[D][34];  // pad 34 to break bank period
  int blk = blockIdx.x;                        // over BATCH*H*(NPAD/BS)
  int mc = blk % (NPAD / BS);
  int h  = (blk / (NPAD / BS)) % H;
  int b  = blk / ((NPAD / BS) * H);
  int o0 = off[b], len = off[b + 1] - o0;
  int t  = threadIdx.x;                        // 256
  int dl = t & 31, ml = t >> 5;                // 8 m-rows x 32 d-cols per step
#pragma unroll
  for (int i = 0; i < 4; ++i) {
#pragma unroll
    for (int c = 0; c < 4; ++c) {
      int m = ml + 8 * i, d = dl + 32 * c;
      int mg = mc * BS + m;
      float x = (mg < len) ? v[(long)(o0 + mg) * HD + h * D + d] : 0.f;
      tile[d][m] = (__bf16)x;
    }
  }
  __syncthreads();
  long obase = (((long)b * H + h) * D) * NPAD + mc * BS;
#pragma unroll
  for (int i = 0; i < 16; ++i) {
    int f = t + 256 * i;     // 0..4095
    int d = f >> 5, m = f & 31;
    vT[obase + (long)d * NPAD + m] = tile[d][m];  // 64B contiguous per 32 lanes
  }
}

// ---------------- K2: block-mean compression ----------------
__global__ void k_compress(const float* __restrict__ k, const float* __restrict__ v,
                           const int* __restrict__ off, __bf16* __restrict__ kcmp,
                           __bf16* __restrict__ vcmpT) {
  int t = blockIdx.x * blockDim.x + threadIdx.x;  // over B*H*NB*D
  int d = t % D;
  int j = (t / D) % NB;
  int h = (t / (D * NB)) % H;
  int b = t / (D * NB * H);
  int o0 = off[b];
  int len = off[b + 1] - o0;
  float ks = 0.f, vs = 0.f;
#pragma unroll 4
  for (int i = 0; i < BS; ++i) {
    int m = j * BS + i;
    if (m < len) {
      long g = (long)(o0 + m) * HD + h * D + d;
      ks += k[g];
      vs += v[g];
    }
  }
  kcmp[(((long)b * H + h) * NB + j) * D + d] = (__bf16)(ks * (1.f / BS));
  vcmpT[(((long)b * H + h) * D + d) * NB + j] = (__bf16)(vs * (1.f / BS));
}

// ---------------- K3: compressed attention + top-8 selection ----------------
__global__ void __launch_bounds__(32) k_cmp_attn(
    const __bf16* __restrict__ qbf, const __bf16* __restrict__ kcmp,
    const __bf16* __restrict__ vcmpT, const float* __restrict__ gcmp,
    const int* __restrict__ off, unsigned* __restrict__ selmask,
    float* __restrict__ ocmp) {
  __shared__ alignas(32) float  sm[16][32];
  __shared__ alignas(32) __bf16 pm[16][32];
  int idx = blockIdx.x;
  int mt = idx % MT;
  int h  = (idx / MT) % H;
  int b  = idx / (MT * H);
  int l = threadIdx.x, half = l >> 4, lq = l & 15;
  const __bf16* qb = qbf   + (((long)b * H + h) * NPAD) * D;
  const __bf16* kc = kcmp  + (((long)b * H + h) * NB) * D;
  const __bf16* vc = vcmpT + (((long)b * H + h) * D) * NB;

  int row = mt * 16 + lq;
  v16bf aq[4];
#pragma unroll
  for (int c = 0; c < 4; ++c) {
    const __bf16* p = qb + (long)row * D + c * 32;
    aq[c] = ld16(p + half * 16, p + half * 16 + 8);
  }
  v8f acc0 = {}, acc1 = {};
#pragma unroll
  for (int c = 0; c < 4; ++c) {
    const __bf16* p0 = kc + (long)lq * D + c * 32 + half * 16;
    const __bf16* p1 = kc + (long)(16 + lq) * D + c * 32 + half * 16;
    acc0 = wmma_bf16(aq[c], *(const v16bf*)p0, acc0);
    acc1 = wmma_bf16(aq[c], *(const v16bf*)p1, acc1);
  }
  // scale + block-causal mask -> LDS (C layout: row = v + 8*half, col = lq)
#pragma unroll
  for (int vv = 0; vv < 8; ++vv) {
    int r = vv + 8 * half;
    int qb_ = (mt * 16 + r) >> 5;
    sm[r][lq]      = (lq <= qb_)        ? acc0[vv] * SCALE : NEGINF;
    sm[r][16 + lq] = ((16 + lq) <= qb_) ? acc1[vv] * SCALE : NEGINF;
  }
  // serial top-8 per row (lanes 0..15; per-wave LDS is in-order -> no barrier)
  if (l < 16) {
    unsigned msk = 0;
#pragma unroll
    for (int s = 0; s < STOPK; ++s) {
      float best = NEGINF;
      int bi = -1;
      for (int j = 0; j < NB; ++j) {
        if (msk & (1u << j)) continue;
        float x = sm[l][j];
        if (x > best) { best = x; bi = j; }
      }
      if (bi >= 0) msk |= 1u << bi;
    }
    selmask[((long)b * H + h) * NPAD + mt * 16 + l] = msk;
  }
  // p_cmp = silu over ALL causal blocks -> bf16 LDS
#pragma unroll
  for (int jj = 0; jj < 16; ++jj) {
    int j = half * 16 + jj;
    float x = sm[lq][j];
    pm[lq][j] = (__bf16)((x == NEGINF) ? 0.f : siluf(x));
  }
  v16bf ap = ld16(&pm[lq][half * 16], &pm[lq][half * 16 + 8]);
  float gv[8];
  long gbase = ((long)b * H + h) * NPAD + mt * 16;
#pragma unroll
  for (int vv = 0; vv < 8; ++vv) gv[vv] = gcmp[gbase + vv + 8 * half];
  int o0 = off[b], len = off[b + 1] - off[b];
#pragma unroll
  for (int t = 0; t < 8; ++t) {
    const __bf16* pv = vc + (long)(t * 16 + lq) * NB + half * 16;
    v8f o = {};
    o = wmma_bf16(ap, *(const v16bf*)pv, o);
#pragma unroll
    for (int vv = 0; vv < 8; ++vv) {
      int mg = mt * 16 + vv + 8 * half;
      if (mg < len)
        ocmp[(long)(o0 + mg) * HD + h * D + t * 16 + lq] = o[vv] * gv[vv];
    }
  }
}

// ---------------- K4: selected attention as masked full attention ----------------
__global__ void __launch_bounds__(32) k_sel_attn(
    const __bf16* __restrict__ qbf, const __bf16* __restrict__ kbf,
    const __bf16* __restrict__ vT, const float* __restrict__ gslc,
    const unsigned* __restrict__ selmask, const int* __restrict__ off,
    float* __restrict__ oslc) {
  __shared__ alignas(32) __bf16 pm[16][32];
  int idx = blockIdx.x;
  int mt = idx % MT;
  int h  = (idx / MT) % H;
  int b  = idx / (MT * H);
  int l = threadIdx.x, half = l >> 4, lq = l & 15;
  const __bf16* qb = qbf + (((long)b * H + h) * NPAD) * D;
  const __bf16* kb = kbf + (((long)b * H + h) * NPAD) * D;
  const __bf16* vb = vT  + (((long)b * H + h) * D) * NPAD;

  int row = mt * 16 + lq;
  v16bf aq[4];
#pragma unroll
  for (int c = 0; c < 4; ++c) {
    const __bf16* p = qb + (long)row * D + c * 32;
    aq[c] = ld16(p + half * 16, p + half * 16 + 8);
  }
  long sbase = ((long)b * H + h) * NPAD + mt * 16;
  unsigned m8[8];
  float gv[8];
  unsigned uni = 0;
#pragma unroll
  for (int vv = 0; vv < 8; ++vv) {
    m8[vv] = selmask[sbase + vv + 8 * half];
    gv[vv] = gslc[sbase + vv + 8 * half];
    uni |= m8[vv];
  }
  uni |= __shfl_xor(uni, 16, 32);  // uniform union of all 16 rows' masks

  v8f oacc[8];
#pragma unroll
  for (int t = 0; t < 8; ++t) oacc[t] = v8f{};

  int jmax = (mt * 16 + 15) >> 5;
  for (int j = 0; j <= jmax; ++j) {
    if (!((uni >> j) & 1)) continue;   // wave-uniform skip
    // prefetch next K block (8KB covered by 32 lanes * 256B) while we compute
    if (j < jmax)
      __builtin_prefetch((const char*)(kb + (long)(j + 1) * BS * D) + l * 256, 0, 3);
    v8f s0 = {}, s1 = {};
#pragma unroll
    for (int c = 0; c < 4; ++c) {
      const __bf16* p0 = kb + (long)(j * BS + lq) * D + c * 32 + half * 16;
      const __bf16* p1 = kb + (long)(j * BS + 16 + lq) * D + c * 32 + half * 16;
      s0 = wmma_bf16(aq[c], *(const v16bf*)p0, s0);
      s1 = wmma_bf16(aq[c], *(const v16bf*)p1, s1);
    }
    // per-row block-selection + elementwise causal mask, silu -> bf16 LDS
#pragma unroll
    for (int vv = 0; vv < 8; ++vv) {
      int r = vv + 8 * half;
      int mg = mt * 16 + r;
      bool sel = (m8[vv] >> j) & 1;
      int kp0 = j * BS + lq;
      pm[r][lq]      = (__bf16)((sel && kp0 <= mg)      ? siluf(s0[vv] * SCALE) : 0.f);
      pm[r][16 + lq] = (__bf16)((sel && kp0 + 16 <= mg) ? siluf(s1[vv] * SCALE) : 0.f);
    }
    v16bf ap = ld16(&pm[lq][half * 16], &pm[lq][half * 16 + 8]);
#pragma unroll
    for (int t = 0; t < 8; ++t) {
      const __bf16* pv = vb + (long)(t * 16 + lq) * NPAD + j * BS + half * 16;
      oacc[t] = wmma_bf16(ap, *(const v16bf*)pv, oacc[t]);
    }
  }
  int o0 = off[b], len = off[b + 1] - off[b];
#pragma unroll
  for (int t = 0; t < 8; ++t) {
#pragma unroll
    for (int vv = 0; vv < 8; ++vv) {
      int mg = mt * 16 + vv + 8 * half;
      if (mg < len)
        oslc[(long)(o0 + mg) * HD + h * D + t * 16 + lq] = oacc[t][vv] * gv[vv];
    }
  }
}

// ---------------- K5: dual LayerNorm * u, summed ----------------
__global__ void k_ln(const float* __restrict__ ocmp, const float* __restrict__ oslc,
                     const float* __restrict__ u, float* __restrict__ out) {
  __shared__ float red[8][4];
  __shared__ float fin[4];
  int t = blockIdx.x;
  int tid = threadIdx.x;
  long base = (long)t * HD;
  float xc[4], xs[4];
  float sc = 0, sc2 = 0, ss = 0, ss2 = 0;
#pragma unroll
  for (int c = 0; c < 4; ++c) {
    int i = tid + 256 * c;
    xc[c] = ocmp[base + i];
    xs[c] = oslc[base + i];
    sc += xc[c]; sc2 += xc[c] * xc[c];
    ss += xs[c]; ss2 += xs[c] * xs[c];
  }
#pragma unroll
  for (int s = 16; s >= 1; s >>= 1) {
    sc += __shfl_xor(sc, s, 32);  sc2 += __shfl_xor(sc2, s, 32);
    ss += __shfl_xor(ss, s, 32);  ss2 += __shfl_xor(ss2, s, 32);
  }
  if ((tid & 31) == 0) {
    int w = tid >> 5;
    red[w][0] = sc; red[w][1] = sc2; red[w][2] = ss; red[w][3] = ss2;
  }
  __syncthreads();
  if (tid < 4) {
    float a = 0;
    for (int w = 0; w < 8; ++w) a += red[w][tid];
    fin[tid] = a;
  }
  __syncthreads();
  float muc = fin[0] * (1.f / HD), mus = fin[2] * (1.f / HD);
  float ic = rsqrtf(fin[1] * (1.f / HD) - muc * muc + 1e-5f);
  float is = rsqrtf(fin[3] * (1.f / HD) - mus * mus + 1e-5f);
#pragma unroll
  for (int c = 0; c < 4; ++c) {
    int i = tid + 256 * c;
    out[base + i] = ((xc[c] - muc) * ic + (xs[c] - mus) * is) * u[base + i];
  }
}

extern "C" void kernel_launch(void* const* d_in, const int* in_sizes, int n_in,
                              void* d_out, int out_size, void* d_ws, size_t ws_size,
                              hipStream_t stream) {
  (void)in_sizes; (void)n_in; (void)out_size; (void)ws_size;
  const float* q  = (const float*)d_in[0];
  const float* k  = (const float*)d_in[1];
  const float* v  = (const float*)d_in[2];
  const float* u  = (const float*)d_in[3];
  const float* Wg = (const float*)d_in[4];
  const float* bg = (const float*)d_in[5];
  const int* off  = (const int*)d_in[6];

  char* ws = (char*)d_ws;
  size_t o = 0;
  const size_t QKV = (size_t)BATCH * H * NPAD * D * sizeof(__bf16);
  __bf16* qbf   = (__bf16*)(ws + o); o += QKV;
  __bf16* kbf   = (__bf16*)(ws + o); o += QKV;
  __bf16* vT    = (__bf16*)(ws + o); o += QKV;
  __bf16* kcmp  = (__bf16*)(ws + o); o += (size_t)BATCH * H * NB * D * sizeof(__bf16);
  __bf16* vcmpT = (__bf16*)(ws + o); o += (size_t)BATCH * H * D * NB * sizeof(__bf16);
  float* gcmp   = (float*)(ws + o);  o += (size_t)BATCH * H * NPAD * sizeof(float);
  float* gslc   = (float*)(ws + o);  o += (size_t)BATCH * H * NPAD * sizeof(float);
  unsigned* sel = (unsigned*)(ws + o); o += (size_t)BATCH * H * NPAD * sizeof(unsigned);
  float* ocmp   = (float*)(ws + o);  o += (size_t)T_TOK * HD * sizeof(float);
  float* oslc   = (float*)(ws + o);  o += (size_t)T_TOK * HD * sizeof(float);

  k_prep<<<(BATCH * NPAD * H) / 8, 256, 0, stream>>>(q, k, Wg, bg, off, qbf, kbf,
                                                     gcmp, gslc);
  k_vtrans<<<BATCH * H * (NPAD / BS), 256, 0, stream>>>(v, off, vT);
  k_compress<<<(BATCH * H * NB * D) / 256, 256, 0, stream>>>(k, v, off, kcmp, vcmpT);
  k_cmp_attn<<<BATCH * H * MT, 32, 0, stream>>>(qbf, kcmp, vcmpT, gcmp, off, sel, ocmp);
  k_sel_attn<<<BATCH * H * MT, 32, 0, stream>>>(qbf, kbf, vT, gslc, sel, off, oslc);
  k_ln<<<T_TOK, 256, 0, stream>>>(ocmp, oslc, u, (float*)d_out);
}
